// SwinBlock_tp_35725537968760
// MI455X (gfx1250) — matrix operand order repaired and checked by reading, not compiled, not run on hardware
//
#include <hip/hip_runtime.h>
#include <hip/hip_bf16.h>

// ---------------------------------------------------------------------------
// Swin block for MI455X (gfx1250): all GEMMs via v_wmma_f32_16x16x32_bf16,
// GEMM tile staging via async global->LDS (ASYNCcnt) with double buffering.
// ---------------------------------------------------------------------------

typedef __attribute__((ext_vector_type(16))) __bf16 v16bf;
typedef __attribute__((ext_vector_type(8)))  __bf16 bf16x8;
typedef __attribute__((ext_vector_type(8)))  float  v8f;

#define B_SZ   32
#define HW     64
#define C_DIM  512
#define NH_    16
#define HD_    32
#define S_TOK  64
#define FF_    2048
#define M_TOK  (B_SZ * HW * HW)   // 131072 tokens
#define SHIFT_ 4

// window-order row -> natural token index (handles roll(-4,-4) + partition,
// and its inverse for the reverse direction — the mapping is identical).
static __device__ __forceinline__ int win_to_tok(int m) {
  int w  = m >> 6;          // window id 0..2047
  int s  = m & 63;          // token in window
  int b  = w >> 6;          // batch
  int wl = w & 63;          // window in image
  int hh = (((wl >> 3) << 3) + (s >> 3) + SHIFT_) & 63;
  int ww = (((wl & 7) << 3) + (s & 7) + SHIFT_) & 63;
  return (b << 12) | (hh << 6) | ww;
}

static __device__ __forceinline__ v16bf pack16(bf16x8 lo, bf16x8 hi) {
  union { v16bf v; bf16x8 h[2]; } u;
  u.h[0] = lo; u.h[1] = hi;
  return u.v;
}

static __device__ __forceinline__ v8f wmma_bf16(v16bf a, v16bf b, v8f c) {
  return __builtin_amdgcn_wmma_f32_16x16x32_bf16(
      false, a, false, b, (short)0, c, false, false);
}

// Async copy of 32 contiguous bytes global -> LDS (2x B128, ASYNCcnt).
// The instruction offset applies to BOTH the LDS and global addresses.
static __device__ __forceinline__ void async_cp32(unsigned lds_off,
                                                  const __bf16* g) {
  asm volatile(
      "global_load_async_to_lds_b128 %0, %1, off\n\t"
      "global_load_async_to_lds_b128 %0, %1, off offset:16"
      :
      : "v"(lds_off), "v"(g)
      : "memory");
}

static __device__ __forceinline__ void async_wait0() {
  asm volatile("s_wait_asynccnt 0x0" ::: "memory");
}

// ---------------------------------------------------------------------------
// Weight transpose + fp32 -> bf16   (src K x N  ->  dst N x K)
// ---------------------------------------------------------------------------
__global__ __launch_bounds__(256) void transpose_cvt(
    const float* __restrict__ src, __bf16* __restrict__ dst, int K, int N) {
  int i = blockIdx.x * 256 + threadIdx.x;
  if (i >= K * N) return;
  int n = i % N, k = i / N;                 // coalesced read along N
  dst[(size_t)n * K + k] = (__bf16)src[i];
}

// ---------------------------------------------------------------------------
// LayerNorm over C=512 per token -> bf16; optional window/roll permutation.
// One block (256 threads) per output row.
// ---------------------------------------------------------------------------
__global__ __launch_bounds__(256) void ln_kernel(
    const float* __restrict__ src, const float* __restrict__ g,
    const float* __restrict__ b, __bf16* __restrict__ dst, int windowed) {
  __shared__ float red[256];
  int m = blockIdx.x;
  int t = threadIdx.x;
  int tok = windowed ? win_to_tok(m) : m;
  const float* x = src + (size_t)tok * C_DIM;
  float v0 = x[t], v1 = x[t + 256];

  red[t] = v0 + v1;
  __syncthreads();
  for (int off = 128; off > 0; off >>= 1) {
    if (t < off) red[t] += red[t + off];
    __syncthreads();
  }
  float mu = red[0] * (1.0f / 512.0f);
  __syncthreads();
  red[t] = v0 * v0 + v1 * v1;
  __syncthreads();
  for (int off = 128; off > 0; off >>= 1) {
    if (t < off) red[t] += red[t + off];
    __syncthreads();
  }
  float var = red[0] * (1.0f / 512.0f) - mu * mu;
  float inv = rsqrtf(var + 1e-5f);

  __bf16* d = dst + (size_t)m * C_DIM;
  d[t]       = (__bf16)((v0 - mu) * inv * g[t] + b[t]);
  d[t + 256] = (__bf16)((v1 - mu) * inv * g[t + 256] + b[t + 256]);
}

// ---------------------------------------------------------------------------
// Tiled bf16 WMMA GEMM: C[M,N] = A[M,K] * Bt[N,K]^T, 128x128 tile / block.
// 256 threads = 8 waves; wave grid 4(m) x 2(n); each wave: 2x4 16x16 tiles.
// Double-buffered async global->LDS staging (prefetch k+1 while computing k).
// MODE epilogues: 0=QKV split(+scale q, transpose v)  1=out-proj(+residual,
// roll-back scatter)  2=MLP1(+GELU)  3=MLP2(+residual -> fp32 out)
// ---------------------------------------------------------------------------
template <int MODE, int NCOLS, int KDIM>
__global__ __launch_bounds__(256) void gemm_wmma(
    const __bf16* __restrict__ A, const __bf16* __restrict__ Bt,
    const float* __restrict__ bias, const float* __restrict__ aux,
    void* __restrict__ out0, void* __restrict__ out1, void* __restrict__ out2) {
  __shared__ __bf16 As[2][128][40];  // pitch 40 -> 16B-aligned, conflict-free
  __shared__ __bf16 Bs[2][128][40];

  const int tid = threadIdx.x;
  const int lane = tid & 31;
  const int wv = tid >> 5;
  const int wm = wv >> 1;          // 0..3
  const int wn = wv & 1;           // 0..1
  const int l15 = lane & 15;
  const int hs  = lane >> 4;       // half-wave select

  const int rowBase = blockIdx.x * 128;
  const int colBase = blockIdx.y * 128;

  const int cpRow = tid >> 1;              // staging: 16 bf16 per thread/tile
  const int cpOff = (tid & 1) << 4;

  const unsigned bufStride = 128u * 40u * 2u;   // bytes per [128][40] buffer
  const unsigned ldsA = (unsigned)(unsigned long long)&As[0][cpRow][cpOff];
  const unsigned ldsB = (unsigned)(unsigned long long)&Bs[0][cpRow][cpOff];
  const __bf16* agBase = A  + (size_t)(rowBase + cpRow) * KDIM + cpOff;
  const __bf16* bgBase = Bt + (size_t)(colBase + cpRow) * KDIM + cpOff;

  v8f acc[2][4];
  v8f zz = {};
#pragma unroll
  for (int i = 0; i < 2; ++i)
#pragma unroll
    for (int j = 0; j < 4; ++j) acc[i][j] = zz;

  const int kbA = hs ? 8 : 0;      // A frag: K runs {kbA..+7, kbA+16..+23}
  const int kbB = hs ? 8 : 0;      // B frag: K runs {kbB..+7, kbB+16..+23}

  // prologue: prefetch first K-tile into buffer 0
  async_cp32(ldsA, agBase);
  async_cp32(ldsB, bgBase);
  async_wait0();
  __syncthreads();

  const int nIter = KDIM / 32;
  for (int it = 0; it < nIter; ++it) {
    const int buf = it & 1;
    if (it + 1 < nIter) {               // prefetch next tile into other buffer
      const int k1 = (it + 1) * 32;
      async_cp32(ldsA + (buf ^ 1) * bufStride, agBase + k1);
      async_cp32(ldsB + (buf ^ 1) * bufStride, bgBase + k1);
    }

    v16bf af[2];
#pragma unroll
    for (int mi = 0; mi < 2; ++mi) {
      int r = wm * 32 + mi * 16 + l15;
      af[mi] = pack16(*(const bf16x8*)&As[buf][r][kbA],
                      *(const bf16x8*)&As[buf][r][kbA + 16]);
    }
#pragma unroll
    for (int ni = 0; ni < 4; ++ni) {
      int c = wn * 64 + ni * 16 + l15;
      v16bf bfr = pack16(*(const bf16x8*)&Bs[buf][c][kbB],
                         *(const bf16x8*)&Bs[buf][c][kbB + 16]);
#pragma unroll
      for (int mi = 0; mi < 2; ++mi)
        acc[mi][ni] = wmma_bf16(af[mi], bfr, acc[mi][ni]);
    }

    async_wait0();                      // next tile landed in LDS
    __syncthreads();                    // everyone done reading current buf
  }

  // epilogue: C/D layout -> lane l15 = N col, vgpr e -> M row (e + hs*8)
#pragma unroll
  for (int mi = 0; mi < 2; ++mi) {
#pragma unroll
    for (int ni = 0; ni < 4; ++ni) {
#pragma unroll
      for (int e = 0; e < 8; ++e) {
        int grow = rowBase + wm * 32 + mi * 16 + hs * 8 + e;
        int gcol = colBase + wn * 64 + ni * 16 + l15;
        float v = acc[mi][ni][e] + bias[gcol];
        if constexpr (MODE == 0) {                 // QKV split
          __bf16* qb = (__bf16*)out0;
          __bf16* kb = (__bf16*)out1;
          __bf16* vT = (__bf16*)out2;
          if (gcol < 512) {
            qb[(size_t)grow * 512 + gcol] = (__bf16)(v * 0.17677669529663687f);
          } else if (gcol < 1024) {
            kb[(size_t)grow * 512 + (gcol - 512)] = (__bf16)v;
          } else {                                 // v stored transposed
            int n = gcol - 1024;
            int d = n & 31, hh = n >> 5;
            int w = grow >> 6, s = grow & 63;
            vT[(((size_t)(w * NH_ + hh)) * HD_ + d) * S_TOK + s] = (__bf16)v;
          }
        } else if constexpr (MODE == 1) {          // out-proj + residual
          float* xo = (float*)out0;
          size_t idx = (size_t)win_to_tok(grow) * C_DIM + gcol;
          xo[idx] = aux[idx] + v;
        } else if constexpr (MODE == 2) {          // MLP1 + tanh-GELU
          float u = v;
          float gl = 0.5f * u *
              (1.0f + tanhf(0.7978845608028654f * (u + 0.044715f * u * u * u)));
          ((__bf16*)out0)[(size_t)grow * FF_ + gcol] = (__bf16)gl;
        } else {                                   // MLP2 + residual -> out
          float* o = (float*)out0;
          size_t idx = (size_t)grow * C_DIM + gcol;
          o[idx] = aux[idx] + v;
        }
      }
    }
  }
}

// ---------------------------------------------------------------------------
// Attention: one block per (window, head); 128 threads = 4 waves.
// scores = q k^T (WMMA, direct global frags), mask+softmax in LDS,
// O = probs @ v (WMMA, probs from LDS, v^T from global).
// ---------------------------------------------------------------------------
static __device__ __forceinline__ int region_label(int wi, int wj, int s) {
  int hh = (wi << 3) + (s >> 3);
  int ww = (wj << 3) + (s & 7);
  int rh = (hh < 56) ? 0 : ((hh < 60) ? 1 : 2);
  int rw = (ww < 56) ? 0 : ((ww < 60) ? 1 : 2);
  return rh * 3 + rw;
}

__global__ __launch_bounds__(128) void attn_kernel(
    const __bf16* __restrict__ q, const __bf16* __restrict__ k,
    const __bf16* __restrict__ vT, __bf16* __restrict__ attn_out) {
  __shared__ float  sc[64][65];     // scores fp32
  __shared__ __bf16 pb[64][72];     // probs bf16, 16B-aligned rows

  const int w = blockIdx.x;         // 0..2047
  const int h = blockIdx.y;         // 0..15
  const int tid = threadIdx.x;
  const int lane = tid & 31;
  const int wv = tid >> 5;          // 0..3 : m-tile
  const int l15 = lane & 15;
  const int hs = lane >> 4;
  const int kbA = hs ? 8 : 0;
  const int kbB = hs ? 8 : 0;

  const __bf16* qw = q  + (size_t)w * S_TOK * C_DIM + h * HD_;
  const __bf16* kw = k  + (size_t)w * S_TOK * C_DIM + h * HD_;
  const __bf16* vw = vT + ((size_t)w * NH_ + h) * HD_ * S_TOK;

  // ---- scores: rows [16*wv, 16*wv+16), all 64 cols; K = HD = 32 (1 wmma)
  v16bf afrag;
  {
    int r = wv * 16 + l15;
    afrag = pack16(*(const bf16x8*)(qw + (size_t)r * C_DIM + kbA),
                   *(const bf16x8*)(qw + (size_t)r * C_DIM + kbA + 16));
  }
#pragma unroll
  for (int j0 = 0; j0 < 4; ++j0) {
    int c = j0 * 16 + l15;
    v16bf bfr = pack16(*(const bf16x8*)(kw + (size_t)c * C_DIM + kbB),
                       *(const bf16x8*)(kw + (size_t)c * C_DIM + kbB + 16));
    v8f s = {};
    s = wmma_bf16(afrag, bfr, s);
#pragma unroll
    for (int e = 0; e < 8; ++e)
      sc[wv * 16 + hs * 8 + e][j0 * 16 + l15] = s[e];
  }
  __syncthreads();

  // ---- mask + softmax: one thread per query row
  if (tid < 64) {
    int wl = w & 63;
    int wi = wl >> 3, wj = wl & 7;
    int labq = region_label(wi, wj, tid);
    float mx = -1e30f;
    for (int j = 0; j < 64; ++j) {
      float v = sc[tid][j];
      if (region_label(wi, wj, j) != labq) v = -10000.0f;
      sc[tid][j] = v;
      mx = fmaxf(mx, v);
    }
    float sum = 0.0f;
    for (int j = 0; j < 64; ++j) {
      float e = __expf(sc[tid][j] - mx);
      sum += e;
      sc[tid][j] = e;
    }
    float inv = 1.0f / sum;
    for (int j = 0; j < 64; ++j) pb[tid][j] = (__bf16)(sc[tid][j] * inv);
  }
  __syncthreads();

  // ---- O = probs @ v : rows 16*wv..+15, cols 0..31; K = 64 (2 steps)
  v8f o[2];
  o[0] = v8f{}; o[1] = v8f{};
#pragma unroll
  for (int ks = 0; ks < 2; ++ks) {
    int r = wv * 16 + l15;
    int ka = ks * 32 + kbA;
    v16bf ap = pack16(*(const bf16x8*)&pb[r][ka],
                      *(const bf16x8*)&pb[r][ka + 16]);
#pragma unroll
    for (int ni = 0; ni < 2; ++ni) {
      int d = ni * 16 + l15;
      int kb = ks * 32 + kbB;
      v16bf bp = pack16(*(const bf16x8*)(vw + (size_t)d * S_TOK + kb),
                        *(const bf16x8*)(vw + (size_t)d * S_TOK + kb + 16));
      o[ni] = wmma_bf16(ap, bp, o[ni]);
    }
  }
#pragma unroll
  for (int ni = 0; ni < 2; ++ni)
#pragma unroll
    for (int e = 0; e < 8; ++e) {
      int row = wv * 16 + hs * 8 + e;
      int col = ni * 16 + l15;
      attn_out[((size_t)w * S_TOK + row) * C_DIM + h * HD_ + col] = (__bf16)o[ni][e];
    }
}

// ---------------------------------------------------------------------------
// Host orchestration
// ---------------------------------------------------------------------------
extern "C" void kernel_launch(void* const* d_in, const int* in_sizes, int n_in,
                              void* d_out, int out_size, void* d_ws, size_t ws_size,
                              hipStream_t stream) {
  const float* hidden = (const float*)d_in[0];
  const float* Wqkv   = (const float*)d_in[1];
  const float* bqkv   = (const float*)d_in[2];
  const float* Wo     = (const float*)d_in[3];
  const float* bo     = (const float*)d_in[4];
  const float* ln1g   = (const float*)d_in[5];
  const float* ln1b   = (const float*)d_in[6];
  const float* ln2g   = (const float*)d_in[7];
  const float* ln2b   = (const float*)d_in[8];
  const float* W1     = (const float*)d_in[9];
  const float* b1     = (const float*)d_in[10];
  const float* W2     = (const float*)d_in[11];
  const float* b2     = (const float*)d_in[12];

  char* ws = (char*)d_ws;
  const size_t MC2  = (size_t)M_TOK * C_DIM * 2;      // 134,217,728 B
  __bf16* wqkvT = (__bf16*)ws;                        // 1536 x 512
  __bf16* woT   = wqkvT + 786432;                     //  512 x 512
  __bf16* w1T   = wqkvT + 1048576;                    // 2048 x 512
  __bf16* w2T   = wqkvT + 2097152;                    //  512 x 2048
  char* base = ws + 6291456;
  __bf16* xw   = (__bf16*)(base);                     // xw -> attn -> ln2 out
  __bf16* qb   = (__bf16*)(base + MC2);
  __bf16* kb   = (__bf16*)(base + 2 * MC2);
  __bf16* vT   = (__bf16*)(base + 3 * MC2);
  __bf16* hmid = qb;                                  // reuse q/k/vT/+1 regions
  float*  xres = (float*)(base + 5 * MC2);            // fp32 residual stream

  // 1) weights -> bf16 transposed
  transpose_cvt<<<(512 * 1536 + 255) / 256, 256, 0, stream>>>(Wqkv, wqkvT, 512, 1536);
  transpose_cvt<<<(512 * 512  + 255) / 256, 256, 0, stream>>>(Wo,   woT,   512, 512);
  transpose_cvt<<<(512 * 2048 + 255) / 256, 256, 0, stream>>>(W1,   w1T,   512, 2048);
  transpose_cvt<<<(2048 * 512 + 255) / 256, 256, 0, stream>>>(W2,   w2T,   2048, 512);

  // 2) LN1 + roll + window partition (bf16 A matrix, window-token order)
  ln_kernel<<<M_TOK, 256, 0, stream>>>(hidden, ln1g, ln1b, xw, 1);

  // 3) QKV GEMM (M=131072, N=1536, K=512)
  gemm_wmma<0, 1536, 512><<<dim3(M_TOK / 128, 12), 256, 0, stream>>>(
      xw, wqkvT, bqkv, nullptr, qb, kb, vT);

  // 4) windowed multi-head attention
  attn_kernel<<<dim3(B_SZ * 64, NH_), 128, 0, stream>>>(qb, kb, vT, xw);

  // 5) out-proj GEMM + roll-back + residual  (N=512, K=512)
  gemm_wmma<1, 512, 512><<<dim3(M_TOK / 128, 4), 256, 0, stream>>>(
      xw, woT, bo, hidden, xres, nullptr, nullptr);

  // 6) LN2
  ln_kernel<<<M_TOK, 256, 0, stream>>>(xres, ln2g, ln2b, xw, 0);

  // 7) MLP1 + GELU (N=2048, K=512)
  gemm_wmma<2, 2048, 512><<<dim3(M_TOK / 128, 16), 256, 0, stream>>>(
      xw, w1T, b1, nullptr, hmid, nullptr, nullptr);

  // 8) MLP2 + residual -> fp32 output (N=512, K=2048)
  gemm_wmma<3, 512, 2048><<<dim3(M_TOK / 128, 4), 256, 0, stream>>>(
      hmid, w2T, b2, xres, (float*)d_out, nullptr, nullptr);
}
